// ResConvBlock_73839077753021
// MI455X (gfx1250) — matrix operand order repaired and checked by reading, not compiled, not run on hardware
//
#include <hip/hip_runtime.h>

typedef _Float16 v16h __attribute__((ext_vector_type(16)));
typedef float    v8f  __attribute__((ext_vector_type(8)));

#define UGRID 175
#define VGRID 50
#define STRIDE_C 8

// ----------------------------------------------------------------------------
// Pack f32 weights (ntaps, cin, cout) into WMMA B-fragment order, f16.
// Fragment f = (k*KT + kt)*NT + nt ; per fragment 32 lanes x 16 halves:
//   lane holds column n = nt*16 + lane%16, K = kt*32 + (lane/16)*8 + (j%8) + (j/8)*16
// ----------------------------------------------------------------------------
__global__ void pack_weights(const float* __restrict__ W, int ntaps, int cin, int cout,
                             _Float16* __restrict__ dst) {
  int KT = cin >> 5, NT = cout >> 4;
  int total = ntaps * KT * NT * 512;
  for (int tid = blockIdx.x * blockDim.x + threadIdx.x; tid < total;
       tid += gridDim.x * blockDim.x) {
    int f = tid >> 9;
    int r = tid & 511;
    int lane = r >> 4;
    int j = r & 15;
    int nt = f % NT;
    int kt = (f / NT) % KT;
    int k = f / (NT * KT);
    int n = nt * 16 + (lane & 15);
    int K = kt * 32 + (lane >> 4) * 8 + (j & 7) + (j >> 3) * 16;
    dst[tid] = (_Float16)W[((long)k * cin + K) * cout + n];
  }
}

// ----------------------------------------------------------------------------
// Gather-conv via v_wmma_f32_16x16x32_f16. One wave per 16-row tile.
// A-fragment (16x32 f16): lane half=lane/16, row m=lane&15:
//   a[0..7]  = feats[row][kt*32 + 8*half + 0..7]
//   a[8..15] = feats[row][kt*32 + 8*half + 16..23]
// C/D: lane holds col n=lane&15, rows 8*half + 0..7 in acc[0..7].
// ----------------------------------------------------------------------------
template <int CIN, int COUT, int NTAPS>
__global__ __launch_bounds__(128) void gconv_wmma(const float* __restrict__ feats,
                                                  const int* __restrict__ nbr,
                                                  const _Float16* __restrict__ wpk,
                                                  int rows, float* __restrict__ out) {
  constexpr int KT = CIN / 32;
  constexpr int NT = COUT / 16;
  const int lane = threadIdx.x & 31;
  const int wave = threadIdx.x >> 5;
  const int tile = blockIdx.x * 4 + wave;
  const int m = lane & 15;
  const int half = lane >> 4;
  const int row = tile * 16 + m;
  const bool rv = row < rows;

  v8f acc[NT];
#pragma unroll
  for (int nt = 0; nt < NT; ++nt)
#pragma unroll
    for (int j = 0; j < 8; ++j) acc[nt][j] = 0.0f;

  for (int k = 0; k < NTAPS; ++k) {
    int idx = rv ? nbr[row * NTAPS + k] : -1;
    const bool ok = idx >= 0;
    const float* src = feats + (long)(ok ? idx : 0) * CIN;
#pragma unroll
    for (int kt = 0; kt < KT; ++kt) {
      const int K0 = kt * 32 + half * 8;
      const float4* s4 = (const float4*)(src + K0);
      float4 f0 = s4[0], f1 = s4[1], f2 = s4[4], f3 = s4[5];
      if (!ok) {
        f0.x = f0.y = f0.z = f0.w = 0.0f;
        f1 = f0; f2 = f0; f3 = f0;
      }
      v16h a;
      a[0] = (_Float16)f0.x;  a[1] = (_Float16)f0.y;  a[2] = (_Float16)f0.z;  a[3] = (_Float16)f0.w;
      a[4] = (_Float16)f1.x;  a[5] = (_Float16)f1.y;  a[6] = (_Float16)f1.z;  a[7] = (_Float16)f1.w;
      a[8] = (_Float16)f2.x;  a[9] = (_Float16)f2.y;  a[10] = (_Float16)f2.z; a[11] = (_Float16)f2.w;
      a[12] = (_Float16)f3.x; a[13] = (_Float16)f3.y; a[14] = (_Float16)f3.z; a[15] = (_Float16)f3.w;
#pragma unroll
      for (int nt = 0; nt < NT; ++nt) {
        const v16h b = *(const v16h*)(wpk + (((k * KT + kt) * NT + nt) * 32 + lane) * 16);
        acc[nt] = __builtin_amdgcn_wmma_f32_16x16x32_f16(false, a, false, b, (short)0,
                                                         acc[nt], false, false);
      }
    }
  }

  const int rbase = tile * 16 + half * 8;
#pragma unroll
  for (int nt = 0; nt < NT; ++nt) {
    const int col = nt * 16 + m;
#pragma unroll
    for (int r = 0; r < 8; ++r) {
      const int rr = rbase + r;
      if (rr < rows) out[(long)rr * COUT + col] = acc[nt][r];
    }
  }
}

// ----------------------------------------------------------------------------
// Deterministic BN statistics: per-block partial sums, then single-block
// ordered finalize producing scale/shift (folds gamma/beta, eps=1e-3).
// ----------------------------------------------------------------------------
__global__ void stats_partial(const float* __restrict__ x, const float* __restrict__ rpre,
                              int rows, int C, float* __restrict__ partial) {
  const int c = threadIdx.x % C;
  const int g = threadIdx.x / C;
  const int groups = blockDim.x / C;
  float s = 0.0f, s2 = 0.0f;
  for (int r = blockIdx.x * groups + g; r < rows; r += gridDim.x * groups) {
    float v = x[(long)r * C + c];
    if (rpre) v += rpre[(long)r * C + c];
    s += v;
    s2 += v * v;
  }
  __shared__ float sh[512];
  sh[threadIdx.x] = s;
  sh[256 + threadIdx.x] = s2;
  __syncthreads();
  if (threadIdx.x < C) {
    const int cc = threadIdx.x;
    float ts = 0.0f, ts2 = 0.0f;
    for (int gg = 0; gg < groups; ++gg) {
      ts += sh[gg * C + cc];
      ts2 += sh[256 + gg * C + cc];
    }
    partial[blockIdx.x * 2 * C + cc] = ts;
    partial[blockIdx.x * 2 * C + C + cc] = ts2;
  }
}

__global__ void stats_finalize(const float* __restrict__ partial, int G, int C, int rows,
                               const float* __restrict__ gamma, const float* __restrict__ beta,
                               float* __restrict__ ss) {
  const int c = threadIdx.x;
  if (c >= C) return;
  float s = 0.0f, s2 = 0.0f;
  for (int g = 0; g < G; ++g) {
    s += partial[g * 2 * C + c];
    s2 += partial[g * 2 * C + C + c];
  }
  const float invn = 1.0f / (float)rows;
  const float mean = s * invn;
  const float var = s2 * invn - mean * mean;
  const float inv = rsqrtf(var + 1e-3f);
  const float sc = gamma[c] * inv;
  ss[c] = sc;
  ss[C + c] = beta[c] - mean * sc;
}

__global__ void bn_apply(const float* __restrict__ x, const float* __restrict__ rpre,
                         const float* __restrict__ rpost, const float* __restrict__ ss,
                         long total, int cmask, int dorelu, float* __restrict__ out) {
  const long stride = (long)gridDim.x * blockDim.x;
  for (long i = (long)blockIdx.x * blockDim.x + threadIdx.x; i < total; i += stride) {
    const int c = (int)(i & (long)cmask);
    float v = x[i];
    if (rpre) v += rpre[i];
    v = v * ss[c] + ss[(cmask + 1) + c];
    if (dorelu) v = fmaxf(v, 0.0f);
    if (rpost) v += rpost[i];
    out[i] = v;
  }
}

// ----------------------------------------------------------------------------
// 2-D projection neighbor map
// ----------------------------------------------------------------------------
__global__ void grid_init(int* __restrict__ grid, int n) {
  for (int i = blockIdx.x * blockDim.x + threadIdx.x; i < n; i += gridDim.x * blockDim.x)
    grid[i] = -1;
}

__global__ void project_kernel(const int* __restrict__ down, const float* __restrict__ l2r,
                               const float* __restrict__ p2, int M, int* __restrict__ uarr,
                               int* __restrict__ varr, int* __restrict__ grid) {
  const int i = blockIdx.x * blockDim.x + threadIdx.x;
  if (i >= M) return;
  const int b = down[4 * i + 0];
  const int z = down[4 * i + 1];
  const int y = down[4 * i + 2];
  const int x = down[4 * i + 3];
  const float vs = 0.05f * (float)STRIDE_C;
  const float fx = (float)x * vs + (0.0f + vs * 0.5f);
  const float fy = (float)y * vs + (-40.0f + vs * 0.5f);
  const float fz = (float)z * vs + (-3.0f + vs * 0.5f);
  const float* L = l2r + b * 12;
  const float r0 = L[0] * fx + L[1] * fy + L[2] * fz + L[3];
  const float r1 = L[4] * fx + L[5] * fy + L[6] * fz + L[7];
  const float r2 = L[8] * fx + L[9] * fy + L[10] * fz + L[11];
  const float* P = p2 + b * 12;
  const float p0 = P[0] * r0 + P[1] * r1 + P[2] * r2 + P[3];
  const float p1 = P[4] * r0 + P[5] * r1 + P[6] * r2 + P[7];
  const float pw = P[8] * r0 + P[9] * r1 + P[10] * r2 + P[11];
  int u = (int)(p0 / pw);   // truncation toward zero, same as astype(int32)
  int v = (int)(p1 / pw);
  u = min(max(u, 0), 1400 - STRIDE_C) / STRIDE_C;
  v = min(max(v, 0), 400 - STRIDE_C) / STRIDE_C;
  uarr[i] = u;
  varr[i] = v;
  atomicMax(&grid[(b * UGRID + u) * VGRID + v], i);
}

__global__ void nbr2d_kernel(const int* __restrict__ down, const int* __restrict__ uarr,
                             const int* __restrict__ varr, const int* __restrict__ grid,
                             int M, int* __restrict__ nbr2d) {
  const int i = blockIdx.x * blockDim.x + threadIdx.x;
  if (i >= M * 9) return;
  const int m = i / 9, j = i % 9;
  const int du = j / 3 - 1, dv = j % 3 - 1;
  const int u = uarr[m] + du, v = varr[m] + dv;
  const int b = down[4 * m];
  nbr2d[i] = (u >= 0 && u < UGRID && v >= 0 && v < VGRID) ? grid[(b * UGRID + u) * VGRID + v] : -1;
}

// ----------------------------------------------------------------------------
extern "C" void kernel_launch(void* const* d_in, const int* in_sizes, int n_in,
                              void* d_out, int out_size, void* d_ws, size_t ws_size,
                              hipStream_t stream) {
  const float* feats   = (const float*)d_in[0];
  const float* Wd1     = (const float*)d_in[1];
  const float* Wres    = (const float*)d_in[2];
  const float* Wsub2   = (const float*)d_in[3];
  const float* W2d1    = (const float*)d_in[4];
  const float* W2d2    = (const float*)d_in[5];
  const float* Wsub3   = (const float*)d_in[6];
  const float* Winv4   = (const float*)d_in[7];
  const float* g32     = (const float*)d_in[8];
  const float* b32     = (const float*)d_in[9];
  const float* g64     = (const float*)d_in[10];
  const float* b64     = (const float*)d_in[11];
  const float* l2r     = (const float*)d_in[12];
  const float* p2      = (const float*)d_in[13];
  const int*   down    = (const int*)d_in[14];
  const int*   nbrdown = (const int*)d_in[15];
  const int*   nbrsub  = (const int*)d_in[16];
  const int*   nbrinv  = (const int*)d_in[17];

  const int B = in_sizes[12] / 12;
  const int N = in_sizes[0] / 64;
  const int M = in_sizes[14] / 4;
  const int G = 256;  // fixed stats grid -> deterministic

  // ---- carve workspace ----
  size_t off = 0;
  auto carve = [&](size_t bytes) -> char* {
    char* p = (char*)d_ws + off;
    off = (off + bytes + 255) & ~(size_t)255;
    return p;
  };
  _Float16* pk_d1   = (_Float16*)carve(108 * 512 * 2);
  _Float16* pk_res  = (_Float16*)carve(108 * 512 * 2);
  _Float16* pk_sub2 = (_Float16*)carve(54 * 512 * 2);
  _Float16* pk_2d1  = (_Float16*)carve(18 * 512 * 2);
  _Float16* pk_2d2  = (_Float16*)carve(18 * 512 * 2);
  _Float16* pk_sub3 = (_Float16*)carve(54 * 512 * 2);
  _Float16* pk_inv4 = (_Float16*)carve(108 * 512 * 2);
  float* B0 = (float*)carve((size_t)N * 64 * 4);  // raw gconv outputs (largest)
  float* B1 = (float*)carve((size_t)M * 32 * 4);
  float* B2 = (float*)carve((size_t)M * 32 * 4);
  float* B3 = (float*)carve((size_t)M * 32 * 4);
  float* B4 = (float*)carve((size_t)M * 32 * 4);
  int* uarr   = (int*)carve((size_t)M * 4);
  int* varr   = (int*)carve((size_t)M * 4);
  int* grid2d = (int*)carve((size_t)B * UGRID * VGRID * 4);
  int* nbr2d  = (int*)carve((size_t)M * 9 * 4);
  float* partial = (float*)carve((size_t)G * 128 * 4);
  float* ss      = (float*)carve(128 * 4);
  float* outf = (float*)d_out;

  auto bn = [&](const float* x, const float* rpre, const float* rpost, const float* gamma,
                const float* beta, int rows, int C, int relu, float* out) {
    stats_partial<<<G, 256, 0, stream>>>(x, rpre, rows, C, partial);
    stats_finalize<<<1, 64, 0, stream>>>(partial, G, C, rows, gamma, beta, ss);
    long total = (long)rows * C;
    bn_apply<<<2048, 256, 0, stream>>>(x, rpre, rpost, ss, total, C - 1, relu, out);
  };

  // ---- pack weights into WMMA B-fragment layout ----
  pack_weights<<<(108 * 512 + 255) / 256, 256, 0, stream>>>(Wd1, 27, 64, 32, pk_d1);
  pack_weights<<<(108 * 512 + 255) / 256, 256, 0, stream>>>(Wres, 27, 64, 32, pk_res);
  pack_weights<<<(54 * 512 + 255) / 256, 256, 0, stream>>>(Wsub2, 27, 32, 32, pk_sub2);
  pack_weights<<<(18 * 512 + 255) / 256, 256, 0, stream>>>(W2d1, 9, 32, 32, pk_2d1);
  pack_weights<<<(18 * 512 + 255) / 256, 256, 0, stream>>>(W2d2, 9, 32, 32, pk_2d2);
  pack_weights<<<(54 * 512 + 255) / 256, 256, 0, stream>>>(Wsub3, 27, 32, 32, pk_sub3);
  pack_weights<<<(108 * 512 + 255) / 256, 256, 0, stream>>>(Winv4, 27, 32, 64, pk_inv4);

  const int blkM = ((M + 15) / 16 + 3) / 4 + 1;
  const int blkN = ((N + 15) / 16 + 3) / 4 + 1;

  // 1) d3 = relu(bn(gconv(features, nbr_down, W_down1)))
  gconv_wmma<64, 32, 27><<<blkM, 128, 0, stream>>>(feats, nbrdown, pk_d1, M, B0);
  bn(B0, nullptr, nullptr, g32 + 0 * 32, b32 + 0 * 32, M, 32, 1, B1);
  // 2) d3 = relu(bn(gconv(d3, nbr_sub, W_sub2)))
  gconv_wmma<32, 32, 27><<<blkM, 128, 0, stream>>>(B1, nbrsub, pk_sub2, M, B0);
  bn(B0, nullptr, nullptr, g32 + 1 * 32, b32 + 1 * 32, M, 32, 1, B2);
  // 3) res = relu(bn(gconv(features, nbr_down, W_res))); d3 = d3 + res  -> B3
  gconv_wmma<64, 32, 27><<<blkM, 128, 0, stream>>>(feats, nbrdown, pk_res, M, B0);
  bn(B0, nullptr, B2, g32 + 2 * 32, b32 + 2 * 32, M, 32, 1, B3);

  // 4) 2-D projection neighbor map
  const int gsz = B * UGRID * VGRID;
  grid_init<<<(gsz + 255) / 256, 256, 0, stream>>>(grid2d, gsz);
  project_kernel<<<(M + 255) / 256, 256, 0, stream>>>(down, l2r, p2, M, uarr, varr, grid2d);
  nbr2d_kernel<<<(M * 9 + 255) / 256, 256, 0, stream>>>(down, uarr, varr, grid2d, M, nbr2d);

  // 5) d2 = relu(bn(gconv(d3, nbr2d, W_2d1)))
  gconv_wmma<32, 32, 9><<<blkM, 128, 0, stream>>>(B3, nbr2d, pk_2d1, M, B0);
  bn(B0, nullptr, nullptr, g32 + 3 * 32, b32 + 3 * 32, M, 32, 1, B1);
  // 6) d2 = relu(bn(gconv(d2, nbr2d, W_2d2)))
  gconv_wmma<32, 32, 9><<<blkM, 128, 0, stream>>>(B1, nbr2d, pk_2d2, M, B0);
  bn(B0, nullptr, nullptr, g32 + 4 * 32, b32 + 4 * 32, M, 32, 1, B2);
  // 7) d3 = bn(d3 + d2)  (stats over the sum; no relu)
  bn(B3, B2, nullptr, g32 + 5 * 32, b32 + 5 * 32, M, 32, 0, B4);
  // 8) d3 = relu(bn(gconv(d3, nbr_sub, W_sub3)))
  gconv_wmma<32, 32, 27><<<blkM, 128, 0, stream>>>(B4, nbrsub, pk_sub3, M, B0);
  bn(B0, nullptr, nullptr, g32 + 6 * 32, b32 + 6 * 32, M, 32, 1, B1);
  // 9) out = relu(bn(gconv(d3, nbr_inv, W_inv4)))  -> d_out
  gconv_wmma<32, 64, 27><<<blkN, 128, 0, stream>>>(B1, nbrinv, pk_inv4, N, B0);
  bn(B0, nullptr, nullptr, g64 + 0 * 64, b64 + 0 * 64, N, 64, 1, outf);
  // 10) out = bn(out + features)  (in-place elementwise; stats first, then apply)
  bn(outf, feats, nullptr, g64 + 1 * 64, b64 + 1 * 64, N, 64, 0, outf);
}
